// SwinTransformer_56513179681250
// MI455X (gfx1250) — matrix operand order repaired and checked by reading, not compiled, not run on hardware
//
#include <hip/hip_runtime.h>
#include <math.h>

typedef __attribute__((ext_vector_type(16))) __bf16 v16bf;
typedef __attribute__((ext_vector_type(8)))  float  v8f;

union FA { v16bf v; unsigned short u[16]; };

__device__ __forceinline__ unsigned short f2bf(float f) {
  union { float f; unsigned u; } x; x.f = f;
  unsigned r = x.u + 0x7FFFu + ((x.u >> 16) & 1u);
  return (unsigned short)(r >> 16);
}

// ---------------- constants ----------------
#define HH 56
#define CC 192
#define NHEAD 6
#define NTOK_W 49
#define NPAD 64
#define NWIN 2048            // B * 64 windows
#define MPAD (NWIN * NPAD)   // 131072 padded window-tokens
#define NTOK 100352          // B * 3136 flat tokens
#define HDIM 32

#define USE_ASYNC_LDS 1

__device__ __forceinline__ unsigned lds_off(const void* p) {
  // flat LDS address: HW uses addr[31:0] as LDS byte address
  return (unsigned)(unsigned long long)p;
}

// ---------------- weight convert ----------------
__global__ void k_convert(const float* __restrict__ in, unsigned short* __restrict__ out, int n) {
  int i = blockIdx.x * 256 + threadIdx.x;
  if (i < n) out[i] = f2bf(in[i]);
}

// ---------------- LN1 + shift + window partition (bf16 out, 64-row padded windows) ----------------
__global__ void k_ln_part(const float* __restrict__ X, const float* __restrict__ g,
                          const float* __restrict__ b, unsigned short* __restrict__ out, int shift) {
  int wid  = (blockIdx.x * blockDim.x + threadIdx.x) >> 5;  // padded window-token
  int lane = threadIdx.x & 31;
  int win = wid >> 6, row = wid & 63;
  unsigned short* o = out + (size_t)wid * CC;
  if (row >= NTOK_W) { for (int c = lane; c < CC; c += 32) o[c] = 0; return; }
  int bb = win >> 6, wimg = win & 63;
  int wh = wimg >> 3, ww = wimg & 7;
  int th = row / 7, tw = row % 7;
  int sh = (wh * 7 + th + shift) % HH;     // roll(-shift) source
  int sw = (ww * 7 + tw + shift) % HH;
  const float* src = X + ((size_t)bb * 3136 + sh * 56 + sw) * CC;
  float vals[6]; float s = 0.f;
  for (int i = 0; i < 6; i++) { vals[i] = src[lane + 32 * i]; s += vals[i]; }
  for (int off = 16; off; off >>= 1) s += __shfl_xor(s, off, 32);
  float mean = s * (1.f / CC);
  float vs = 0.f;
  for (int i = 0; i < 6; i++) { float d = vals[i] - mean; vs += d * d; }
  for (int off = 16; off; off >>= 1) vs += __shfl_xor(vs, off, 32);
  float rstd = rsqrtf(vs * (1.f / CC) + 1e-5f);
  for (int i = 0; i < 6; i++) {
    int c = lane + 32 * i;
    o[c] = f2bf((vals[i] - mean) * rstd * g[c] + b[c]);
  }
}

// ---------------- LN2 flat ----------------
__global__ void k_ln_flat(const float* __restrict__ X, const float* __restrict__ g,
                          const float* __restrict__ b, unsigned short* __restrict__ out) {
  int wid  = (blockIdx.x * blockDim.x + threadIdx.x) >> 5;
  int lane = threadIdx.x & 31;
  const float* src = X + (size_t)wid * CC;
  unsigned short* o = out + (size_t)wid * CC;
  float vals[6]; float s = 0.f;
  for (int i = 0; i < 6; i++) { vals[i] = src[lane + 32 * i]; s += vals[i]; }
  for (int off = 16; off; off >>= 1) s += __shfl_xor(s, off, 32);
  float mean = s * (1.f / CC);
  float vs = 0.f;
  for (int i = 0; i < 6; i++) { float d = vals[i] - mean; vs += d * d; }
  for (int off = 16; off; off >>= 1) vs += __shfl_xor(vs, off, 32);
  float rstd = rsqrtf(vs * (1.f / CC) + 1e-5f);
  for (int i = 0; i < 6; i++) {
    int c = lane + 32 * i;
    o[c] = f2bf((vals[i] - mean) * rstd * g[c] + b[c]);
  }
}

// ---------------- 256x64 block GEMM core (bf16 -> f32 acc), wave tile 32x64 ----------------
#define LPA 40   // LDS row stride (bf16) for A tile [256][32]
#define LPB 40   // LDS row stride (bf16) for transposed B tile [64][32]

struct Acc { v8f t[2][4]; };

__device__ __forceinline__ void gemm_block(const unsigned short* __restrict__ A, int lda, int m0,
                                           const unsigned short* __restrict__ W, int ldw, int n0,
                                           int K, unsigned short* sA, unsigned short* sB, Acc& acc) {
  const int tid = threadIdx.x, lane = tid & 31, wid = tid >> 5;
  const int wm0 = wid * 32;                 // 8 waves -> 8 M strips of 32
  const int mrow = lane & 15, half = lane >> 4;
  for (int i = 0; i < 2; i++) for (int j = 0; j < 4; j++)
    for (int e = 0; e < 8; e++) acc.t[i][j][e] = 0.f;

  for (int k0 = 0; k0 < K; k0 += 32) {
    // stage A 256x32 (4 chunks of 8 bf16 per thread)
    for (int c = tid; c < 1024; c += 256) {
      int r = c >> 2, cc = (c & 3) << 3;
      const unsigned short* src = A + (size_t)(m0 + r) * lda + k0 + cc;
#if USE_ASYNC_LDS
      unsigned dst = lds_off(&sA[r * LPA + cc]);
      asm volatile("global_load_async_to_lds_b128 %0, %1, off"
                   :: "v"(dst), "v"((unsigned long long)src) : "memory");
#else
      *(uint4*)&sA[r * LPA + cc] = *(const uint4*)src;
#endif
      if (k0 + 32 < K) __builtin_prefetch(src + 32, 0, 0);
    }
    // stage W 32x64 transposed -> sB[n][k]
    {
      int c = tid;                               // 256 chunks of 8
      int k = c >> 3, nn = (c & 7) << 3;
      const unsigned short* src = W + (size_t)(k0 + k) * ldw + n0 + nn;
      uint4 d = *(const uint4*)src;
      unsigned short tmp[8]; *(uint4*)tmp = d;
      for (int q = 0; q < 8; q++) sB[(nn + q) * LPB + k] = tmp[q];
      if (k0 + 32 < K) __builtin_prefetch(src + (size_t)32 * ldw, 0, 0);
    }
#if USE_ASYNC_LDS
    asm volatile("s_wait_asynccnt 0" ::: "memory");
#endif
    __syncthreads();
    FA a[2], b[4];
    for (int mt = 0; mt < 2; mt++) {
      const unsigned short* p = &sA[(wm0 + mt * 16 + mrow) * LPA];
      for (int j = 0; j < 8; j++) { a[mt].u[j] = p[half * 8 + j]; a[mt].u[8 + j] = p[16 + half * 8 + j]; }
    }
    for (int nt = 0; nt < 4; nt++) {
      const unsigned short* p = &sB[(nt * 16 + mrow) * LPB];
      for (int j = 0; j < 8; j++) { b[nt].u[j] = p[half * 8 + j]; b[nt].u[8 + j] = p[16 + half * 8 + j]; }
    }
    for (int mt = 0; mt < 2; mt++)
      for (int nt = 0; nt < 4; nt++)
        acc.t[mt][nt] = __builtin_amdgcn_wmma_f32_16x16x32_bf16(
            false, a[mt].v, false, b[nt].v, (short)0, acc.t[mt][nt], false, false);
    __syncthreads();
  }
}

// element (mt,nt,r) maps to row m0 + wid*32 + mt*16 + (lane>>4)*8 + r, col n0 + nt*16 + (lane&15)

// ---------------- QKV GEMM + split/scale ----------------
__global__ void k_qkv(const unsigned short* __restrict__ A, const unsigned short* __restrict__ W,
                      const float* __restrict__ bias, unsigned short* __restrict__ Q,
                      unsigned short* __restrict__ Kb, unsigned short* __restrict__ V) {
  __shared__ unsigned short sA[256 * LPA], sB[64 * LPB];
  int m0 = blockIdx.y * 256, n0 = blockIdx.x * 64;
  Acc acc;
  gemm_block(A, CC, m0, W, 3 * CC, n0, CC, sA, sB, acc);
  int lane = threadIdx.x & 31, wid = threadIdx.x >> 5;
  const float scale = 0.17677669529663687f;  // 32^-0.5
  for (int mt = 0; mt < 2; mt++) for (int nt = 0; nt < 4; nt++) for (int r = 0; r < 8; r++) {
    int m = m0 + wid * 32 + mt * 16 + ((lane >> 4) << 3) + r;
    int n = n0 + nt * 16 + (lane & 15);
    float v = acc.t[mt][nt][r] + bias[n];
    int which = n / CC, head = (n % CC) >> 5, d = n & 31;
    int win = m >> 6, row = m & 63;
    size_t idx = (((size_t)win * NHEAD + head) * NPAD + row) * HDIM + d;
    if (which == 0) Q[idx] = f2bf(v * scale);
    else if (which == 1) Kb[idx] = f2bf(v);
    else V[idx] = f2bf(v);
  }
}

// ---------------- attention: S=QK^T (+bias,+mask), softmax, P@V ----------------
__global__ void k_attn(const unsigned short* __restrict__ Qb, const unsigned short* __restrict__ Kb,
                       const unsigned short* __restrict__ Vb, const float* __restrict__ bt,
                       unsigned short* __restrict__ out, int shifted) {
  __shared__ float S[64 * 65];
  __shared__ unsigned short P[64 * 72];
  __shared__ unsigned short Vt[32 * 72];        // V transposed: Vt[d][k]
  int win = blockIdx.x, head = blockIdx.y;
  const unsigned short* Q = Qb + ((size_t)win * NHEAD + head) * NPAD * HDIM;
  const unsigned short* Km = Kb + ((size_t)win * NHEAD + head) * NPAD * HDIM;
  const unsigned short* V  = Vb + ((size_t)win * NHEAD + head) * NPAD * HDIM;
  int tid = threadIdx.x, lane = tid & 31, w = tid >> 5;
  int mrow = lane & 15, half = lane >> 4;
  // stage V (64x32) transposed into LDS: each thread reads 16 contiguous (half a V row)
  {
    int k = tid >> 1, d0 = (tid & 1) * 16;
    const unsigned short* vp = V + k * HDIM + d0;
    for (int j = 0; j < 16; j++) Vt[(d0 + j) * 72 + k] = vp[j];
  }
  // A fragment: Q rows [w*16, w*16+16), K-dim = 32 in one chunk
  FA a;
  { const unsigned short* qp = Q + (w * 16 + mrow) * HDIM;
    for (int j = 0; j < 8; j++) { a.u[j] = qp[half * 8 + j]; a.u[8 + j] = qp[16 + half * 8 + j]; } }
  int wimg = win & 63, wh = wimg >> 3, ww = wimg & 7;
  for (int nt = 0; nt < 4; nt++) {
    FA b;
    { const unsigned short* kp = Km + (nt * 16 + mrow) * HDIM;
      for (int j = 0; j < 8; j++) { b.u[j] = kp[half * 8 + j]; b.u[8 + j] = kp[16 + half * 8 + j]; } }
    v8f c; for (int e = 0; e < 8; e++) c[e] = 0.f;
    c = __builtin_amdgcn_wmma_f32_16x16x32_bf16(false, a.v, false, b.v, (short)0, c, false, false);
    for (int r = 0; r < 8; r++) {
      int i = w * 16 + half * 8 + r;
      int j = nt * 16 + mrow;
      float v = c[r];
      if (j >= NTOK_W) v = -1e30f;
      else if (i < NTOK_W) {
        int ih = i / 7, iw = i % 7, jh = j / 7, jw = j % 7;
        int rel = (ih - jh + 6) * 13 + (iw - jw + 6);
        v += bt[rel * NHEAD + head];
        if (shifted) {
          int gih = wh * 7 + ih, giw = ww * 7 + iw, gjh = wh * 7 + jh, gjw = ww * 7 + jw;
          int ri = ((gih < 49) ? 0 : (gih < 53) ? 1 : 2) * 3 + ((giw < 49) ? 0 : (giw < 53) ? 1 : 2);
          int rj = ((gjh < 49) ? 0 : (gjh < 53) ? 1 : 2) * 3 + ((gjw < 49) ? 0 : (gjw < 53) ? 1 : 2);
          if (ri != rj) v -= 100.f;
        }
      }
      S[i * 65 + j] = v;
    }
  }
  __syncthreads();
  if (tid < 64) {
    float* row = &S[tid * 65];
    float mx = -1e30f;
    for (int j = 0; j < NTOK_W; j++) mx = fmaxf(mx, row[j]);
    float e[NTOK_W]; float sum = 0.f;
    for (int j = 0; j < NTOK_W; j++) { e[j] = __expf(row[j] - mx); sum += e[j]; }
    float inv = 1.f / sum;
    unsigned short* pr = &P[tid * 72];
    for (int j = 0; j < NTOK_W; j++) pr[j] = f2bf(e[j] * inv);
    for (int j = NTOK_W; j < 64; j++) pr[j] = 0;
  }
  __syncthreads();
  v8f o[2];
  for (int nt = 0; nt < 2; nt++) for (int e = 0; e < 8; e++) o[nt][e] = 0.f;
  for (int kc = 0; kc < 64; kc += 32) {
    FA pa;
    { const unsigned short* pp = &P[(w * 16 + mrow) * 72 + kc];
      for (int j = 0; j < 8; j++) { pa.u[j] = pp[half * 8 + j]; pa.u[8 + j] = pp[16 + half * 8 + j]; } }
    for (int nt = 0; nt < 2; nt++) {
      FA vb;
      int n = nt * 16 + mrow, kb = kc;
      const unsigned short* vp = &Vt[n * 72 + kb];
      for (int j = 0; j < 8; j++) { vb.u[j] = vp[half * 8 + j]; vb.u[8 + j] = vp[16 + half * 8 + j]; }
      o[nt] = __builtin_amdgcn_wmma_f32_16x16x32_bf16(false, pa.v, false, vb.v, (short)0, o[nt], false, false);
    }
  }
  for (int nt = 0; nt < 2; nt++) for (int r = 0; r < 8; r++) {
    int i = w * 16 + half * 8 + r;
    int d = nt * 16 + mrow;
    out[((size_t)win * NPAD + i) * CC + head * HDIM + d] = f2bf(o[nt][r]);
  }
}

// ---------------- proj GEMM + window reverse + unshift + residual ----------------
__global__ void k_proj(const unsigned short* __restrict__ A, const unsigned short* __restrict__ W,
                       const float* __restrict__ bias, const float* __restrict__ Xin,
                       float* __restrict__ Xout, int shift) {
  __shared__ unsigned short sA[256 * LPA], sB[64 * LPB];
  int m0 = blockIdx.y * 256, n0 = blockIdx.x * 64;
  Acc acc;
  gemm_block(A, CC, m0, W, CC, n0, CC, sA, sB, acc);
  int lane = threadIdx.x & 31, wid = threadIdx.x >> 5;
  for (int mt = 0; mt < 2; mt++) for (int nt = 0; nt < 4; nt++) for (int r = 0; r < 8; r++) {
    int m = m0 + wid * 32 + mt * 16 + ((lane >> 4) << 3) + r;
    int n = n0 + nt * 16 + (lane & 15);
    int win = m >> 6, row = m & 63;
    if (row >= NTOK_W) continue;
    int bb = win >> 6, wimg = win & 63;
    int wh = wimg >> 3, ww = wimg & 7;
    int th = row / 7, tw = row % 7;
    int dh = (wh * 7 + th + shift) % HH;   // roll(+shift) destination
    int dw = (ww * 7 + tw + shift) % HH;
    size_t tok = (size_t)bb * 3136 + dh * 56 + dw;
    Xout[tok * CC + n] = Xin[tok * CC + n] + acc.t[mt][nt][r] + bias[n];
  }
}

// ---------------- FC1 GEMM + GELU (exact erf) ----------------
__global__ void k_fc1(const unsigned short* __restrict__ A, const unsigned short* __restrict__ W,
                      const float* __restrict__ bias, unsigned short* __restrict__ out) {
  __shared__ unsigned short sA[256 * LPA], sB[64 * LPB];
  int m0 = blockIdx.y * 256, n0 = blockIdx.x * 64;
  Acc acc;
  gemm_block(A, CC, m0, W, 4 * CC, n0, CC, sA, sB, acc);
  int lane = threadIdx.x & 31, wid = threadIdx.x >> 5;
  for (int mt = 0; mt < 2; mt++) for (int nt = 0; nt < 4; nt++) for (int r = 0; r < 8; r++) {
    int m = m0 + wid * 32 + mt * 16 + ((lane >> 4) << 3) + r;
    int n = n0 + nt * 16 + (lane & 15);
    float v = acc.t[mt][nt][r] + bias[n];
    float gv = 0.5f * v * (1.f + erff(v * 0.70710678118654752f));
    out[(size_t)m * (4 * CC) + n] = f2bf(gv);
  }
}

// ---------------- FC2 GEMM + residual ----------------
__global__ void k_fc2(const unsigned short* __restrict__ A, const unsigned short* __restrict__ W,
                      const float* __restrict__ bias, const float* __restrict__ Xin,
                      float* __restrict__ Xout) {
  __shared__ unsigned short sA[256 * LPA], sB[64 * LPB];
  int m0 = blockIdx.y * 256, n0 = blockIdx.x * 64;
  Acc acc;
  gemm_block(A, 4 * CC, m0, W, CC, n0, 4 * CC, sA, sB, acc);
  int lane = threadIdx.x & 31, wid = threadIdx.x >> 5;
  for (int mt = 0; mt < 2; mt++) for (int nt = 0; nt < 4; nt++) for (int r = 0; r < 8; r++) {
    int m = m0 + wid * 32 + mt * 16 + ((lane >> 4) << 3) + r;
    int n = n0 + nt * 16 + (lane & 15);
    Xout[(size_t)m * CC + n] = Xin[(size_t)m * CC + n] + acc.t[mt][nt][r] + bias[n];
  }
}

// ---------------- launch ----------------
extern "C" void kernel_launch(void* const* d_in, const int* in_sizes, int n_in,
                              void* d_out, int out_size, void* d_ws, size_t ws_size,
                              hipStream_t stream) {
  const float* x       = (const float*)d_in[0];
  const float* n1g     = (const float*)d_in[1];
  const float* n1b     = (const float*)d_in[2];
  const float* qkv_w   = (const float*)d_in[3];
  const float* qkv_b   = (const float*)d_in[4];
  const float* bt      = (const float*)d_in[5];
  const float* proj_w  = (const float*)d_in[6];
  const float* proj_b  = (const float*)d_in[7];
  const float* n2g     = (const float*)d_in[8];
  const float* n2b     = (const float*)d_in[9];
  const float* fc1_w   = (const float*)d_in[10];
  const float* fc1_b   = (const float*)d_in[11];
  const float* fc2_w   = (const float*)d_in[12];
  const float* fc2_b   = (const float*)d_in[13];

  char* ws = (char*)d_ws;
  const size_t oX = 0;                               // fp32 residual stream  77,070,336 B
  const size_t oA = oX + 77070336;                   // bf16 activations      50,331,648 B
  const size_t oB = oA + 50331648;                   // bf16 QKV / MLP hidden 154,140,672 B
  const size_t oWq = oB + 154140672;                 // bf16 weights
  const size_t oWp = oWq + 2ull * 110592 * 2;
  const size_t oW1 = oWp + 2ull * 36864 * 2;
  const size_t oW2 = oW1 + 2ull * 147456 * 2;

  float* bufX = (float*)(ws + oX);
  unsigned short* bufA = (unsigned short*)(ws + oA);
  unsigned short* bufB = (unsigned short*)(ws + oB);
  unsigned short* wq = (unsigned short*)(ws + oWq);
  unsigned short* wp = (unsigned short*)(ws + oWp);
  unsigned short* w1 = (unsigned short*)(ws + oW1);
  unsigned short* w2 = (unsigned short*)(ws + oW2);

  unsigned short* Q  = bufB;
  unsigned short* Kb = bufB + 25165824;
  unsigned short* V  = bufB + 50331648;

  k_convert<<<(221184 + 255) / 256, 256, 0, stream>>>(qkv_w, wq, 221184);
  k_convert<<<(73728 + 255) / 256, 256, 0, stream>>>(proj_w, wp, 73728);
  k_convert<<<(294912 + 255) / 256, 256, 0, stream>>>(fc1_w, w1, 294912);
  k_convert<<<(294912 + 255) / 256, 256, 0, stream>>>(fc2_w, w2, 294912);

  for (int d = 0; d < 2; d++) {
    int shift = (d == 0) ? 0 : 3;
    const float* Xcur = (d == 0) ? x : bufX;

    k_ln_part<<<MPAD / 8, 256, 0, stream>>>(Xcur, n1g + d * CC, n1b + d * CC, bufA, shift);

    dim3 g2(9, MPAD / 256);
    k_qkv<<<g2, 256, 0, stream>>>(bufA, wq + (size_t)d * 110592, qkv_b + d * 3 * CC, Q, Kb, V);

    dim3 g3(NWIN, NHEAD);
    k_attn<<<g3, 128, 0, stream>>>(Q, Kb, V, bt + (size_t)d * 169 * NHEAD, bufA, shift ? 1 : 0);

    dim3 g4(3, MPAD / 256);
    k_proj<<<g4, 256, 0, stream>>>(bufA, wp + (size_t)d * 36864, proj_b + d * CC, Xcur, bufX, shift);

    k_ln_flat<<<NTOK / 8, 256, 0, stream>>>(bufX, n2g + d * CC, n2b + d * CC, bufA);

    dim3 g6(12, NTOK / 256);
    k_fc1<<<g6, 256, 0, stream>>>(bufA, w1 + (size_t)d * 147456, fc1_b + d * 4 * CC, bufB);

    dim3 g7(3, NTOK / 256);
    float* outp = (d == 1) ? (float*)d_out : bufX;
    k_fc2<<<g7, 256, 0, stream>>>(bufB, w2 + (size_t)d * 147456, fc2_b + d * CC, bufX, outp);
  }
}
